// CDs2Hg_57552561766661
// MI455X (gfx1250) — matrix operand order repaired and checked by reading, not compiled, not run on hardware
//
#include <hip/hip_runtime.h>
#include <hip/hip_bf16.h>

// ---------------- problem constants ----------------
#define NROWS   16384      // N samples
#define NCLS    2048       // C classes
#define FEAT    128        // F embed
#define KDIM    256        // 2F packed GEMM depth
#define TAUC    32.0f
#define ALPHAC  0.9f

typedef __bf16  v16bf __attribute__((ext_vector_type(16)));
typedef __bf16  v8bf  __attribute__((ext_vector_type(8)));
typedef float   v8f   __attribute__((ext_vector_type(8)));

// LDS row stride for B panels (bf16 elems): 264*2B=528B=132 dwords, 132%64=4
// -> 16 rows land on 16 distinct banks; 528%16==0 keeps b128 alignment.
#define BSTR 264

__device__ __forceinline__ float wave_sum32(float v) {
    v += __shfl_xor(v, 16, 32);
    v += __shfl_xor(v,  8, 32);
    v += __shfl_xor(v,  4, 32);
    v += __shfl_xor(v,  2, 32);
    v += __shfl_xor(v,  1, 32);
    return v;
}

__device__ __forceinline__ void split_bf16(float x, __bf16* hi, __bf16* lo) {
    __bf16 h = (__bf16)x;
    *hi = h;
    *lo = (__bf16)(x - (float)h);
}

__device__ __forceinline__ v16bf frag16(const __bf16* p0, const __bf16* p1) {
    v8bf a = *(const v8bf*)p0;
    v8bf b = *(const v8bf*)p1;
    v16bf r;
#pragma unroll
    for (int i = 0; i < 8; ++i) { r[i] = a[i]; r[8 + i] = b[i]; }
    return r;
}

// ---------------- K1: build A = [Xn^2 | Xn], bf16 hi/lo ----------------
__global__ void prep_x_kernel(const float* __restrict__ X,
                              __bf16* __restrict__ Ah, __bf16* __restrict__ Al) {
    int gwave = (blockIdx.x * blockDim.x + threadIdx.x) >> 5;  // one wave per row
    int lane  = threadIdx.x & 31;
    if (gwave >= NROWS) return;
    const float4 v = ((const float4*)(X + (size_t)gwave * FEAT))[lane];
    float ss = v.x * v.x + v.y * v.y + v.z * v.z + v.w * v.w;
    ss = wave_sum32(ss);
    float inv = 1.0f / fmaxf(sqrtf(ss), 1e-12f);
    float xn[4] = { v.x * inv, v.y * inv, v.z * inv, v.w * inv };
    int base = gwave * KDIM + lane * 4;
#pragma unroll
    for (int j = 0; j < 4; ++j) {
        __bf16 h, l;
        float sq = xn[j] * xn[j];
        split_bf16(sq, &h, &l);
        Ah[base + j] = h;            Al[base + j] = l;
        split_bf16(xn[j], &h, &l);
        Ah[base + FEAT + j] = h;     Al[base + FEAT + j] = l;
    }
}

// ------- K2: build B = [inv_cov | -2*mu*inv_cov], bf16 hi/lo, + constC -------
__global__ void prep_c_kernel(const float* __restrict__ means,
                              const float* __restrict__ log_vars,
                              __bf16* __restrict__ Bh, __bf16* __restrict__ Bl,
                              float* __restrict__ constC) {
    int gwave = (blockIdx.x * blockDim.x + threadIdx.x) >> 5;  // one wave per class
    int lane  = threadIdx.x & 31;
    if (gwave >= NCLS) return;
    const float4 m  = ((const float4*)(means    + (size_t)gwave * FEAT))[lane];
    const float4 lv = ((const float4*)(log_vars + (size_t)gwave * FEAT))[lane];
    float mm[4] = { m.x, m.y, m.z, m.w };
    float lvv[4] = { lv.x, lv.y, lv.z, lv.w };
    float ic[4];
    float ss = 0.0f;
#pragma unroll
    for (int j = 0; j < 4; ++j) {
        ic[j] = __expf(-fminf(fmaxf(lvv[j], 0.0f), 6.0f));   // exp(-relu6(lv))
        ss += mm[j] * mm[j];
    }
    ss = wave_sum32(ss);
    float inv = 1.0f / fmaxf(sqrtf(ss), 1e-12f);
    int base = gwave * KDIM + lane * 4;
    float cpart = 0.0f;
#pragma unroll
    for (int j = 0; j < 4; ++j) {
        float mu = mm[j] * inv;
        cpart += mu * mu * ic[j];
        __bf16 h, l;
        split_bf16(ic[j], &h, &l);
        Bh[base + j] = h;          Bl[base + j] = l;
        split_bf16(-2.0f * mu * ic[j], &h, &l);
        Bh[base + FEAT + j] = h;   Bl[base + FEAT + j] = l;
    }
    cpart = wave_sum32(cpart);
    if (lane == 0) constC[gwave] = cpart;
}

// ---------------- K3: fused bf16x3 WMMA GEMM + epilogue ----------------
// Block: 256 threads (8 waves). Tile: 128 rows x 64 cols. K = 256.
// grid = (NCLS/64 = 32, NROWS/128 = 128)
__global__ void gemm_fused_kernel(const __bf16* __restrict__ Ah,
                                  const __bf16* __restrict__ Al,
                                  const __bf16* __restrict__ Bh,
                                  const __bf16* __restrict__ Bl,
                                  const float* __restrict__ constC,
                                  const int* __restrict__ T,
                                  float* __restrict__ H,
                                  float* __restrict__ Dt,
                                  float* __restrict__ Spart) {
    extern __shared__ __bf16 smem[];
    __bf16* sBh = smem;                 // 64 x BSTR
    __bf16* sBl = smem + 64 * BSTR;     // 64 x BSTR

    const int cb   = blockIdx.x;                    // column block (0..31)
    const int rb   = blockIdx.y;                    // row block    (0..127)
    const int tid  = threadIdx.x;
    const int wave = tid >> 5;
    const int lane = tid & 31;
    const int hf   = lane >> 4;                     // half-wave (0/1)
    const int r    = lane & 15;
    const int n0   = cb * 64;
    const int m0   = rb * 128 + wave * 16;

    // Stage B panels (64 classes x 256 K) into LDS, hi and lo.
    for (int idx = tid; idx < 64 * 32; idx += 256) {
        int row = idx >> 5;
        int c8  = (idx & 31) * 8;
        *(v8bf*)(sBh + row * BSTR + c8) = *(const v8bf*)(Bh + (size_t)(n0 + row) * KDIM + c8);
        *(v8bf*)(sBl + row * BSTR + c8) = *(const v8bf*)(Bl + (size_t)(n0 + row) * KDIM + c8);
    }
    __syncthreads();

    v8f acc[4];
#pragma unroll
    for (int t = 0; t < 4; ++t)
#pragma unroll
        for (int j = 0; j < 8; ++j) acc[t][j] = 0.0f;

    const __bf16* arh = Ah + (size_t)(m0 + r) * KDIM;
    const __bf16* arl = Al + (size_t)(m0 + r) * KDIM;

#pragma unroll
    for (int ks = 0; ks < KDIM; ks += 32) {
        // A fragment (16x32 bf16): elems 0..7 = K(8h..), elems 8..15 = K(16+8h..)
        v16bf fah = frag16(arh + ks + 8 * hf, arh + ks + 16 + 8 * hf);
        v16bf fal = frag16(arl + ks + 8 * hf, arl + ks + 16 + 8 * hf);
#pragma unroll
        for (int t = 0; t < 4; ++t) {
            const __bf16* bp = sBh + (t * 16 + r) * BSTR + ks + 8 * hf;
            const __bf16* bq = sBl + (t * 16 + r) * BSTR + ks + 8 * hf;
            v16bf fbh = frag16(bp, bp + 16);
            v16bf fbl = frag16(bq, bq + 16);
            // D += Ah*Bh + Ah*Bl + Al*Bh  (fp32 emulation, drop lo*lo)
            acc[t] = __builtin_amdgcn_wmma_f32_16x16x32_bf16(
                false, fah, false, fbh, (short)0, acc[t], false, false);
            acc[t] = __builtin_amdgcn_wmma_f32_16x16x32_bf16(
                false, fah, false, fbl, (short)0, acc[t], false, false);
            acc[t] = __builtin_amdgcn_wmma_f32_16x16x32_bf16(
                false, fal, false, fbh, (short)0, acc[t], false, false);
        }
    }

    // Epilogue: D = acc + constC; H = (c==T[n]) ? 1 : exp(-alpha*D);
    // per-row partial sum of exp(-tau*D); target distance Dt.
#pragma unroll
    for (int j = 0; j < 8; ++j) {
        const int row  = m0 + j + 8 * hf;          // C/D layout: M = j + 8*half
        const int tcls = T[row];
        float s = 0.0f;
#pragma unroll
        for (int t = 0; t < 4; ++t) {
            const int col = n0 + t * 16 + r;
            const float D = acc[t][j] + constC[col];
            float Hv;
            if (col == tcls) {
                Hv = 1.0f;
                Dt[row] = D;                        // exactly one writer grid-wide
            } else {
                Hv = __expf(-ALPHAC * D);
            }
            H[(size_t)row * NCLS + col] = Hv;
            s += __expf(-TAUC * D);                 // D>=0 => no max-shift needed
        }
        // reduce the 16 lanes of this half-wave (rows are per-half)
        s += __shfl_xor(s, 1, 32);
        s += __shfl_xor(s, 2, 32);
        s += __shfl_xor(s, 4, 32);
        s += __shfl_xor(s, 8, 32);
        if (r == 0) Spart[row * 32 + cb] = s;       // deterministic plain store
    }
}

// ---------------- K4: loss = mean( log(sum_c e^{-tau D}) + tau*D_t ) ---------
__global__ void loss_kernel(const float* __restrict__ Spart,
                            const float* __restrict__ Dt,
                            float* __restrict__ out) {
    __shared__ float red[8];
    float local = 0.0f;
    for (int n = threadIdx.x; n < NROWS; n += 256) {
        float s = 0.0f;
#pragma unroll
        for (int c = 0; c < 32; ++c) s += Spart[n * 32 + c];
        local += logf(s) + TAUC * Dt[n];
    }
    local = wave_sum32(local);
    int wave = threadIdx.x >> 5, lane = threadIdx.x & 31;
    if (lane == 0) red[wave] = local;
    __syncthreads();
    if (threadIdx.x == 0) {
        float t = 0.0f;
#pragma unroll
        for (int w = 0; w < 8; ++w) t += red[w];
        out[0] = t / (float)NROWS;
    }
}

// ---------------- launcher ----------------
extern "C" void kernel_launch(void* const* d_in, const int* in_sizes, int n_in,
                              void* d_out, int out_size, void* d_ws, size_t ws_size,
                              hipStream_t stream) {
    const float* X        = (const float*)d_in[0];
    const int*   T        = (const int*)  d_in[1];
    const float* means    = (const float*)d_in[2];
    const float* log_vars = (const float*)d_in[3];

    float* out  = (float*)d_out;        // out[0] = loss_mean
    float* Hout = out + 1;              // out[1..] = H [N, C]

    // workspace carving (16B aligned sections)
    char* ws = (char*)d_ws;
    __bf16* Ah     = (__bf16*)(ws);                                   // 8 MB
    __bf16* Al     = (__bf16*)(ws + (size_t)8  * 1024 * 1024);        // 8 MB
    __bf16* Bh     = (__bf16*)(ws + (size_t)16 * 1024 * 1024);        // 1 MB
    __bf16* Bl     = (__bf16*)(ws + (size_t)17 * 1024 * 1024);        // 1 MB
    float*  constC = (float*) (ws + (size_t)18 * 1024 * 1024);        // 8 KB
    float*  Dt     = (float*) (ws + (size_t)18 * 1024 * 1024 + 65536);   // 64 KB
    float*  Spart  = (float*) (ws + (size_t)18 * 1024 * 1024 + 131072);  // 2 MB

    // K1: 8 rows/block (one wave each)
    prep_x_kernel<<<NROWS / 8, 256, 0, stream>>>(X, Ah, Al);
    // K2: 8 classes/block
    prep_c_kernel<<<NCLS / 8, 256, 0, stream>>>(means, log_vars, Bh, Bl, constC);
    // K3: fused GEMM; dyn LDS = 2 panels * 64 * BSTR * 2B
    dim3 grid(NCLS / 64, NROWS / 128);
    size_t smem = (size_t)2 * 64 * BSTR * sizeof(__bf16);
    gemm_fused_kernel<<<grid, 256, smem, stream>>>(Ah, Al, Bh, Bl, constC, T,
                                                   Hout, Dt, Spart);
    // K4: scalar loss
    loss_kernel<<<1, 256, 0, stream>>>(Spart, Dt, out);
}
